// AttentionBlock_6425271075384
// MI455X (gfx1250) — matrix operand order repaired and checked by reading, not compiled, not run on hardware
//
#include <hip/hip_runtime.h>
#include <hip/hip_bf16.h>
#include <math.h>

// ---------------------------------------------------------------------------
// Types for gfx1250 WMMA (wave32). bf16 A/B fragments (16x32 / 32x16), f32 acc.
// ---------------------------------------------------------------------------
typedef __bf16 bf16_t;
typedef __attribute__((ext_vector_type(16))) __bf16 v16bf;
typedef __attribute__((ext_vector_type(8)))  __bf16 v8bf;
typedef __attribute__((ext_vector_type(8)))  float  v8f;
typedef __attribute__((ext_vector_type(4)))  unsigned int v4u;
typedef __attribute__((ext_vector_type(8)))  int v8i;
typedef __attribute__((ext_vector_type(4)))  int v4i;

#define T_SEQ    2048
#define HIDDEN   2880
#define HEAD_DIM 64
#define N_HEADS  64
#define N_KV     8
#define QKV_DIM  5120          // 64*(64+16)
#define QD       4096          // N_HEADS*HEAD_DIM
#define KD       512           // N_KV*HEAD_DIM
#define SM_SCALE 0.125f        // 1/sqrt(64)

// ---------------------------------------------------------------------------
// Fragment loaders (layouts per CDNA5 ISA 7.12.2, wave32)
// A 16x32 bf16: lane L<16 holds row M=L, K = {0..7, 16..23}; L>=16: row M=L-16,
// K = {8..15, 24..31}. Two contiguous 16-byte loads per lane.
// ---------------------------------------------------------------------------
__device__ __forceinline__ v16bf load_a_frag(const bf16_t* __restrict__ A,
                                             int lda, int row0, int k0) {
  int lane = threadIdx.x & 31;
  int half = lane >> 4;
  int r    = lane & 15;
  const bf16_t* p = A + (size_t)(row0 + r) * lda + k0 + half * 8;
  v8bf lo = *(const v8bf*)p;
  v8bf hi = *(const v8bf*)(p + 16);
  v16bf out;
#pragma unroll
  for (int i = 0; i < 8; ++i) { out[i] = lo[i]; out[8 + i] = hi[i]; }
  return out;
}

// Same A-fragment gather but from an LDS tile (row stride 64 elements).
__device__ __forceinline__ v16bf lds_a_frag(const bf16_t* Ab /*[rows][64]*/,
                                            int row0, int k0) {
  int lane = threadIdx.x & 31;
  int half = lane >> 4;
  int r    = lane & 15;
  const bf16_t* p = Ab + (row0 + r) * 64 + k0 + half * 8;
  v8bf lo = *(const v8bf*)p;
  v8bf hi = *(const v8bf*)(p + 16);
  v16bf out;
#pragma unroll
  for (int i = 0; i < 8; ++i) { out[i] = lo[i]; out[8 + i] = hi[i]; }
  return out;
}

// B 32x16 bf16 (K x N): lane holds column n = col0 + (lane&15), K values
// {0..7, 16..23} + (lane>>4)*8 (same order as A).  With W stored row-major
// (N, K) this is two 16-byte reads of a weight row.
__device__ __forceinline__ v16bf load_b_frag(const bf16_t* __restrict__ B,
                                             int ldb, int col0, int k0) {
  int lane = threadIdx.x & 31;
  int n  = col0 + (lane & 15);
  int ks = k0 + (lane >> 4) * 8;
  const bf16_t* p = B + (size_t)n * ldb + ks;
  v8bf lo = *(const v8bf*)p;
  v8bf hi = *(const v8bf*)(p + 16);
  v16bf out;
#pragma unroll
  for (int i = 0; i < 8; ++i) { out[i] = lo[i]; out[8 + i] = hi[i]; }
  return out;
}

__device__ __forceinline__ v8f wmma_bf16(v16bf a, v16bf b, v8f c) {
  return __builtin_amdgcn_wmma_f32_16x16x32_bf16(
      /*neg_a=*/false, a, /*neg_b=*/false, b,
      /*c_mod=*/(short)0, c, /*reuse_a=*/false, /*reuse_b=*/false);
}

// ---------------------------------------------------------------------------
// TDM: async 2D tile load Global -> LDS via tensor descriptor (ISA ch.8).
// Tensor dims are set equal to tile dims (tile always fully in-bounds), so the
// OOB/clip path is never exercised. data_size=1 => 2-byte elements.
// ---------------------------------------------------------------------------
__device__ __forceinline__ void tdm_load_2d(unsigned lds_off, const bf16_t* gaddr,
                                            unsigned tile_rows, unsigned tile_cols,
                                            unsigned row_stride_elems) {
  unsigned long long ga = (unsigned long long)gaddr;
  v4u g0;
  g0[0] = 1u;                                    // count=1, user mode, no gather
  g0[1] = lds_off;                               // lds_addr (bytes)
  g0[2] = (unsigned)(ga & 0xFFFFFFFFu);          // global_addr[31:0]
  g0[3] = (unsigned)((ga >> 32) & 0x01FFFFFFu)   // global_addr[56:32]
        | (2u << 30);                            // type = 2 ("image")
  v8i g1;
  g1[0] = (int)(1u << 16);                       // data_size=1 (2B); mask=0
  g1[1] = (int)((tile_cols & 0xFFFFu) << 16);    // tensor_dim0[15:0] @ bits 63:48
  g1[2] = (int)(((tile_cols >> 16) & 0xFFFFu)    // tensor_dim0[31:16]
        | ((tile_rows & 0xFFFFu) << 16));        // tensor_dim1[15:0]
  g1[3] = (int)(((tile_rows >> 16) & 0xFFFFu)    // tensor_dim1[31:16]
        | ((tile_cols & 0xFFFFu) << 16));        // tile_dim0
  g1[4] = (int)(tile_rows & 0xFFFFu);            // tile_dim1 (tile_dim2 = 0)
  g1[5] = (int)row_stride_elems;                 // tensor_dim0_stride[31:0]
  g1[6] = 0;                                     // stride hi + dim1_stride lo
  g1[7] = 0;
  v4i z = {0, 0, 0, 0};
#if __clang_major__ >= 23
  v8i z8 = {0, 0, 0, 0, 0, 0, 0, 0};
  __builtin_amdgcn_tensor_load_to_lds(g0, g1, z, z, z8, 0);
#else
  __builtin_amdgcn_tensor_load_to_lds(g0, g1, z, z, 0);
#endif
}

// ---------------------------------------------------------------------------
// 1) RMSNorm -> bf16 activations
// ---------------------------------------------------------------------------
__global__ void rmsnorm_to_bf16(const float* __restrict__ x,
                                const float* __restrict__ scale,
                                bf16_t* __restrict__ out) {
  int row = blockIdx.x;
  const float* xr = x + (size_t)row * HIDDEN;
  __shared__ float red[256];
  float ss = 0.f;
  for (int c = threadIdx.x; c < HIDDEN; c += 256) { float v = xr[c]; ss += v * v; }
  red[threadIdx.x] = ss;
  __syncthreads();
  for (int s = 128; s > 0; s >>= 1) {
    if (threadIdx.x < s) red[threadIdx.x] += red[threadIdx.x + s];
    __syncthreads();
  }
  float inv = rsqrtf(red[0] * (1.0f / HIDDEN) + 1e-5f);
  for (int c = threadIdx.x; c < HIDDEN; c += 256)
    out[(size_t)row * HIDDEN + c] = (bf16_t)(xr[c] * inv * scale[c]);
}

// ---------------------------------------------------------------------------
// 2) f32 -> bf16 weight conversion
// ---------------------------------------------------------------------------
__global__ void f32_to_bf16(const float* __restrict__ in, bf16_t* __restrict__ out,
                            int n) {
  int i = blockIdx.x * 256 + threadIdx.x;
  if (i < n) out[i] = (bf16_t)in[i];
}

// ---------------------------------------------------------------------------
// 3) WMMA GEMM: C[M,N] = A[M,K] * W[N,K]^T + bias (+ resid).
//    Block tile 128(M) x 64(N); 4 waves, each a 32x64 slice (8 WMMA / k-step).
//    A panel (128 x 64 per chunk) staged into double-buffered LDS by the TDM,
//    issued one chunk ahead by wave 0; B streams from global (L2-resident)
//    with a near-cache prefetch hint one chunk ahead.
//    __launch_bounds__(128, 1): lift the VGPR cap so the 8-accumulator tile
//    plus the pipelined fragment loads never spill to scratch.
// ---------------------------------------------------------------------------
__global__ void __launch_bounds__(128, 1)
gemm_bf16(const bf16_t* __restrict__ A, const bf16_t* __restrict__ W,
          const float* __restrict__ bias,
          const float* __restrict__ resid,
          float* __restrict__ C, int K, int ldc) {
  __shared__ bf16_t Atile[2][128][64];   // 32 KB double buffer

  int wave = threadIdx.x >> 5;
  int lane = threadIdx.x & 31;
  int bm = blockIdx.y * 128;             // block row base
  int n0 = blockIdx.x * 64;              // block col base (all waves share N)
  int m0 = wave * 32;                    // wave row base within tile

  v8f acc[2][4] = {};
  const bf16_t* Ag = A + (size_t)bm * K;
  int nchunks = K >> 6;                  // K / 64

  if (wave == 0) {
    tdm_load_2d((unsigned)(size_t)&Atile[0][0][0], Ag, 128, 64, (unsigned)K);
    __builtin_amdgcn_s_wait_tensorcnt(0);
  }
  __syncthreads();

  for (int c = 0; c < nchunks; ++c) {
    if (wave == 0 && (c + 1) < nchunks) {
      tdm_load_2d((unsigned)(size_t)&Atile[(c + 1) & 1][0][0],
                  Ag + (size_t)(c + 1) * 64, 128, 64, (unsigned)K);
    }
    // prefetch next B chunk; locality 3 -> keep in near caches for reuse
    if ((c + 1) < nchunks) {
      int n = n0 + (lane & 15);
      __builtin_prefetch(W + (size_t)n * K + (c + 1) * 64 + (lane >> 4) * 16, 0, 3);
    }

    const bf16_t* Ab = &Atile[c & 1][0][0];
    int kg = c * 64;
#pragma unroll
    for (int ks = 0; ks < 64; ks += 32) {
      v16bf A0 = lds_a_frag(Ab, m0,      ks);
      v16bf A1 = lds_a_frag(Ab, m0 + 16, ks);
      v16bf B0 = load_b_frag(W, K, n0,      kg + ks);
      v16bf B1 = load_b_frag(W, K, n0 + 16, kg + ks);
      v16bf B2 = load_b_frag(W, K, n0 + 32, kg + ks);
      v16bf B3 = load_b_frag(W, K, n0 + 48, kg + ks);
      acc[0][0] = wmma_bf16(A0, B0, acc[0][0]);
      acc[0][1] = wmma_bf16(A0, B1, acc[0][1]);
      acc[0][2] = wmma_bf16(A0, B2, acc[0][2]);
      acc[0][3] = wmma_bf16(A0, B3, acc[0][3]);
      acc[1][0] = wmma_bf16(A1, B0, acc[1][0]);
      acc[1][1] = wmma_bf16(A1, B1, acc[1][1]);
      acc[1][2] = wmma_bf16(A1, B2, acc[1][2]);
      acc[1][3] = wmma_bf16(A1, B3, acc[1][3]);
    }
    if (wave == 0) __builtin_amdgcn_s_wait_tensorcnt(0);
    __syncthreads();
  }

  int nn = lane & 15;
  int mb = (lane >> 4) * 8;
#pragma unroll
  for (int i = 0; i < 2; ++i) {
#pragma unroll
    for (int j = 0; j < 4; ++j) {
      int n = n0 + j * 16 + nn;
      float bv = bias[n];
#pragma unroll
      for (int r = 0; r < 8; ++r) {
        int m = bm + m0 + i * 16 + mb + r;
        float v = acc[i][j][r] + bv;
        if (resid) v += resid[(size_t)m * ldc + n];
        C[(size_t)m * ldc + n] = v;
      }
    }
  }
}

// ---------------------------------------------------------------------------
// 4) RoPE (YaRN-interp) + pack:
//    Qh (H, T, 64) bf16 row-major; Kh (G, T, 64) bf16; Vt (G, 64, T) bf16.
// ---------------------------------------------------------------------------
__global__ void rope_pack(const float* __restrict__ qkv,
                          bf16_t* __restrict__ Qh, bf16_t* __restrict__ Kh,
                          bf16_t* __restrict__ Vt) {
  int t = blockIdx.x;
  const float* row = qkv + (size_t)t * QKV_DIM;
  __shared__ float cs[32], sn[32];
  if (threadIdx.x < 32) {
    int i = threadIdx.x;
    const float TWO_PI = 6.28318530717958647692f;
    float logb = logf(150000.0f);
    float freq = __expf((2.0f * (float)i / (float)HEAD_DIM) * logb);
    float conc = 0.1f * logf(32.0f) + 1.0f;
    float d_half = 32.0f;
    float low  = d_half * logf(4096.0f / (32.0f * TWO_PI)) / logb;
    float high = d_half * logf(4096.0f / (1.0f  * TWO_PI)) / logb;
    float interp = 1.0f / (32.0f * freq);
    float extrap = 1.0f / freq;
    float ramp = ((float)i - low) / (high - low);
    float mask = 1.0f - fminf(fmaxf(ramp, 0.0f), 1.0f);
    float invf = interp * (1.0f - mask) + extrap * mask;
    float ang = (float)t * invf;
    cs[i] = cosf(ang) * conc;
    sn[i] = sinf(ang) * conc;
  }
  __syncthreads();
  for (int idx = threadIdx.x; idx < N_HEADS * 32; idx += blockDim.x) {
    int h = idx >> 5, i = idx & 31;
    float x1 = row[h * 64 + i], x2 = row[h * 64 + i + 32];
    bf16_t* dst = Qh + ((size_t)h * T_SEQ + t) * 64;
    dst[i]      = (bf16_t)(x1 * cs[i] - x2 * sn[i]);
    dst[i + 32] = (bf16_t)(x2 * cs[i] + x1 * sn[i]);
  }
  for (int idx = threadIdx.x; idx < N_KV * 32; idx += blockDim.x) {
    int g = idx >> 5, i = idx & 31;
    const float* kr = row + QD + g * 64;
    float x1 = kr[i], x2 = kr[i + 32];
    bf16_t* dst = Kh + ((size_t)g * T_SEQ + t) * 64;
    dst[i]      = (bf16_t)(x1 * cs[i] - x2 * sn[i]);
    dst[i + 32] = (bf16_t)(x2 * cs[i] + x1 * sn[i]);
  }
  for (int idx = threadIdx.x; idx < N_KV * 64; idx += blockDim.x) {
    int g = idx >> 6, d = idx & 63;
    Vt[((size_t)g * 64 + d) * T_SEQ + t] = (bf16_t)row[QD + KD + g * 64 + d];
  }
}

// ---------------------------------------------------------------------------
// 5) Causal GQA flash attention. One wave per (head, 16-query tile).
//    S = Q K^T via WMMA; online softmax through per-wave LDS tile; O += P V.
// ---------------------------------------------------------------------------
__global__ void __launch_bounds__(128, 1)
attention(const bf16_t* __restrict__ Qh, const bf16_t* __restrict__ Kh,
          const bf16_t* __restrict__ Vt, bf16_t* __restrict__ Ao) {
  __shared__ float S[4][16][32];
  __shared__ float rowScale[4][16];

  int wave = threadIdx.x >> 5;
  int lane = threadIdx.x & 31;
  int half = lane >> 4;
  int nsub = lane & 15;

  int h = blockIdx.y;
  int g = h >> 3;
  int q0 = (blockIdx.x * 4 + wave) * 16;

  const bf16_t* Q = Qh + (size_t)h * T_SEQ * 64;
  const bf16_t* K = Kh + (size_t)g * T_SEQ * 64;
  const bf16_t* V = Vt + (size_t)g * 64 * T_SEQ;

  v16bf qa0 = load_a_frag(Q, 64, q0, 0);
  v16bf qa1 = load_a_frag(Q, 64, q0, 32);

  v8f o0 = {}, o1 = {}, o2 = {}, o3 = {};
  float m_run = -3e38f, l_run = 0.f;

  float (*Sw)[32] = S[wave];
  float* scW = rowScale[wave];

  int kend = q0 + 16;
  for (int j0 = 0; j0 < kend; j0 += 32) {
#pragma unroll
    for (int sub = 0; sub < 2; ++sub) {
      int kbase = j0 + sub * 16;
      v16bf kb0 = load_b_frag(K, 64, kbase, 0);
      v16bf kb1 = load_b_frag(K, 64, kbase, 32);
      v8f s = {};
      s = wmma_bf16(qa0, kb0, s);
      s = wmma_bf16(qa1, kb1, s);
      int kidx = kbase + nsub;
#pragma unroll
      for (int r = 0; r < 8; ++r) {
        int qidx = q0 + r + 8 * half;
        float v = s[r] * SM_SCALE;
        if (kidx > qidx) v = -1e30f;
        Sw[r + 8 * half][sub * 16 + nsub] = v;
      }
    }
    asm volatile("s_wait_dscnt 0x0" ::: "memory");

    if (lane < 16) {
      float rmax = -3e38f;
#pragma unroll
      for (int c = 0; c < 32; ++c) rmax = fmaxf(rmax, Sw[lane][c]);
      float nm = fmaxf(m_run, rmax);
      float corr = __expf(m_run - nm);
      float rsum = 0.f;
#pragma unroll
      for (int c = 0; c < 32; ++c) {
        float p = __expf(Sw[lane][c] - nm);
        Sw[lane][c] = p;
        rsum += p;
      }
      l_run = l_run * corr + rsum;
      m_run = nm;
      scW[lane] = corr;
    }
    asm volatile("s_wait_dscnt 0x0" ::: "memory");

#pragma unroll
    for (int r = 0; r < 8; ++r) {
      float c = scW[r + 8 * half];
      o0[r] *= c; o1[r] *= c; o2[r] *= c; o3[r] *= c;
    }

    v16bf pa;
    {
      int mrow = lane & 15;
      int cb = half * 8;
#pragma unroll
      for (int i = 0; i < 8; ++i) {
        pa[i]     = (bf16_t)Sw[mrow][cb + i];
        pa[8 + i] = (bf16_t)Sw[mrow][cb + 16 + i];
      }
    }

    v16bf vb0 = load_b_frag(V, T_SEQ, 0,  j0);
    v16bf vb1 = load_b_frag(V, T_SEQ, 16, j0);
    v16bf vb2 = load_b_frag(V, T_SEQ, 32, j0);
    v16bf vb3 = load_b_frag(V, T_SEQ, 48, j0);
    o0 = wmma_bf16(pa, vb0, o0);
    o1 = wmma_bf16(pa, vb1, o1);
    o2 = wmma_bf16(pa, vb2, o2);
    o3 = wmma_bf16(pa, vb3, o3);
  }

  if (lane < 16) scW[lane] = 1.0f / l_run;
  asm volatile("s_wait_dscnt 0x0" ::: "memory");
#pragma unroll
  for (int r = 0; r < 8; ++r) {
    float inv = scW[r + 8 * half];
    int q = q0 + r + 8 * half;
    bf16_t* dst = Ao + (size_t)q * QD + h * 64;
    dst[nsub]      = (bf16_t)(o0[r] * inv);
    dst[16 + nsub] = (bf16_t)(o1[r] * inv);
    dst[32 + nsub] = (bf16_t)(o2[r] * inv);
    dst[48 + nsub] = (bf16_t)(o3[r] * inv);
  }
}

// ---------------------------------------------------------------------------
// Launch
// ---------------------------------------------------------------------------
extern "C" void kernel_launch(void* const* d_in, const int* in_sizes, int n_in,
                              void* d_out, int out_size, void* d_ws, size_t ws_size,
                              hipStream_t stream) {
  const float* x          = (const float*)d_in[0];
  const float* norm_scale = (const float*)d_in[1];
  const float* qkv_w      = (const float*)d_in[2];
  const float* qkv_b      = (const float*)d_in[3];
  const float* out_w      = (const float*)d_in[4];
  const float* out_b      = (const float*)d_in[5];
  float* out = (float*)d_out;

  char* ws = (char*)d_ws;
  size_t off = 0;
  auto carve = [&](size_t bytes) -> void* {
    void* p = ws + off;
    off += (bytes + 255) & ~(size_t)255;
    return p;
  };
  bf16_t* t_bf    = (bf16_t*)carve((size_t)T_SEQ * HIDDEN * sizeof(bf16_t));
  bf16_t* wqkv_bf = (bf16_t*)carve((size_t)QKV_DIM * HIDDEN * sizeof(bf16_t));
  bf16_t* wout_bf = (bf16_t*)carve((size_t)HIDDEN * QD * sizeof(bf16_t));
  float*  qkv_f   = (float*)carve((size_t)T_SEQ * QKV_DIM * sizeof(float));
  bf16_t* Qh      = (bf16_t*)carve((size_t)N_HEADS * T_SEQ * HEAD_DIM * sizeof(bf16_t));
  bf16_t* Kh      = (bf16_t*)carve((size_t)N_KV * T_SEQ * HEAD_DIM * sizeof(bf16_t));
  bf16_t* Vt      = (bf16_t*)carve((size_t)N_KV * HEAD_DIM * T_SEQ * sizeof(bf16_t));
  bf16_t* Ao      = (bf16_t*)carve((size_t)T_SEQ * QD * sizeof(bf16_t));

  // 1) RMSNorm -> bf16
  rmsnorm_to_bf16<<<T_SEQ, 256, 0, stream>>>(x, norm_scale, t_bf);

  // 2) weight conversions
  {
    int n1 = QKV_DIM * HIDDEN;
    f32_to_bf16<<<(n1 + 255) / 256, 256, 0, stream>>>(qkv_w, wqkv_bf, n1);
    int n2 = HIDDEN * QD;
    f32_to_bf16<<<(n2 + 255) / 256, 256, 0, stream>>>(out_w, wout_bf, n2);
  }

  // 3) QKV GEMM: (2048 x 2880) x (2880 -> 5120)
  gemm_bf16<<<dim3(QKV_DIM / 64, T_SEQ / 128), 128, 0, stream>>>(
      t_bf, wqkv_bf, qkv_b, /*resid=*/nullptr, qkv_f, HIDDEN, QKV_DIM);

  // 4) RoPE + pack Q/K/V
  rope_pack<<<T_SEQ, 256, 0, stream>>>(qkv_f, Qh, Kh, Vt);

  // 5) causal GQA attention
  attention<<<dim3(T_SEQ / 64, N_HEADS), 128, 0, stream>>>(Qh, Kh, Vt, Ao);

  // 6) output projection + bias + residual
  gemm_bf16<<<dim3(HIDDEN / 64, T_SEQ / 128), 128, 0, stream>>>(
      Ao, wout_bf, out_b, /*resid=*/x, out, QD, HIDDEN);
}
